// LatentSlotAttention_86887188398343
// MI455X (gfx1250) — matrix operand order repaired and checked by reading, not compiled, not run on hardware
//
#include <hip/hip_runtime.h>
#include <math.h>

typedef __attribute__((ext_vector_type(16))) __bf16 v16bf;
typedef __attribute__((ext_vector_type(8)))  float  v8f;

#define LN_EPS 1e-5f
#define ATT_SCALE 0.17677669529663687f  // 32^-0.5
#define WAVES_PER_WG 4
#define WAVE_SCR 1696  // ks(512)+vs(512)+slots(128)+scr(512)+stats(32)

__device__ __forceinline__ v8f wmma_bf16(v16bf a, v16bf b, v8f c) {
  // D = A(16x32 bf16) * B(32x16 bf16) + C(16x16 f32)
  return __builtin_amdgcn_wmma_f32_16x16x32_bf16(
      /*neg_a=*/false, a, /*neg_b=*/false, b,
      /*c_mod=*/(short)0, c, /*reuse_a=*/false, /*reuse_b=*/false);
}

// B-operand tile stored in LDS pre-packed: tile[j*32+lane] = {B[kb+2j][n], B[kb+2j+1][n]}
// with n = lane&15, kb = (lane>>4)*16  (contiguous k = kb + s, same order as packA).
__device__ __forceinline__ v16bf loadB(const unsigned* tile, int lane) {
  v16bf b;
#pragma unroll
  for (int j = 0; j < 8; ++j) {
    union { unsigned u; __bf16 h[2]; } c;
    c.u = tile[j * 32 + lane];
    b[2 * j] = c.h[0];
    b[2 * j + 1] = c.h[1];
  }
  return b;
}

// A-operand (16x32 bf16): lane m=L&15; lanes<16 hold K 0..15,
// lanes>=16 hold K 16..31 (pairs per VGPR).
template <typename F>
__device__ __forceinline__ v16bf packA(int lane, F get) {
  v16bf a;
  int m = lane & 15;
  int kb = (lane >> 4) * 16;
#pragma unroll
  for (int j = 0; j < 4; ++j) {
    a[2 * j]     = (__bf16)get(m, kb + 2 * j);
    a[2 * j + 1] = (__bf16)get(m, kb + 2 * j + 1);
    a[8 + 2 * j]     = (__bf16)get(m, kb + 8 + 2 * j);
    a[8 + 2 * j + 1] = (__bf16)get(m, kb + 8 + 2 * j + 1);
  }
  return a;
}

__device__ __forceinline__ float sigm(float x) { return 1.f / (1.f + __expf(-x)); }

// tiles: [0,2) w_in  [2,4) wk  [4,6) wv  [6,8) wq  [8,14) wih  [14,20) whh
__global__ void __launch_bounds__(128, 1)
slot_attn_kernel(const float* __restrict__ latent, const float* __restrict__ slot_mu,
                 const float* __restrict__ w_in, const float* __restrict__ b_in,
                 const float* __restrict__ ln_g, const float* __restrict__ ln_b,
                 const float* __restrict__ wq, const float* __restrict__ wk,
                 const float* __restrict__ wv, const float* __restrict__ wih,
                 const float* __restrict__ whh, const float* __restrict__ bih,
                 const float* __restrict__ bhh, float* __restrict__ out, int Bn) {
  __shared__ unsigned sWB[20 * 256];          // bf16x2-packed B tiles (20 KB)
  __shared__ float sBias[288];                // b_in | ln_g | ln_b | bih | bhh
  __shared__ float sMu[128];                  // slot_mu
  __shared__ float sWave[WAVES_PER_WG][WAVE_SCR];

  const int tid = threadIdx.x;
  const int lane = tid & 31;
  const int wid = tid >> 5;

  // ---- cooperative weight pack into WMMA B-layout (bf16 pairs) ----
  for (int e = tid; e < 20 * 256; e += 128) {
    int t = e >> 8, r = e & 255;
    int j = r >> 5, ln = r & 31;
    int n = ln & 15, kb = (ln >> 4) << 4;
    int k0 = kb + 2 * j, k1 = k0 + 1;
    float w0, w1;
    if (t < 2)       { int c = n + 16 * t;        w0 = (k0 < 8) ? w_in[c * 8 + k0] : 0.f;
                                                  w1 = (k1 < 8) ? w_in[c * 8 + k1] : 0.f; }
    else if (t < 4)  { int c = n + 16 * (t - 2);  w0 = wk[c * 32 + k0];  w1 = wk[c * 32 + k1]; }
    else if (t < 6)  { int c = n + 16 * (t - 4);  w0 = wv[c * 32 + k0];  w1 = wv[c * 32 + k1]; }
    else if (t < 8)  { int c = n + 16 * (t - 6);  w0 = wq[c * 32 + k0];  w1 = wq[c * 32 + k1]; }
    else if (t < 14) { int c = n + 16 * (t - 8);  w0 = wih[c * 32 + k0]; w1 = wih[c * 32 + k1]; }
    else             { int c = n + 16 * (t - 14); w0 = whh[c * 32 + k0]; w1 = whh[c * 32 + k1]; }
    union { unsigned u; __bf16 h[2]; } cv;
    cv.h[0] = (__bf16)w0; cv.h[1] = (__bf16)w1;
    sWB[t * 256 + j * 32 + ln] = cv.u;
  }
  for (int i = tid; i < 288; i += 128) {
    float v;
    if (i < 32) v = b_in[i];
    else if (i < 64) v = ln_g[i - 32];
    else if (i < 96) v = ln_b[i - 64];
    else if (i < 192) v = bih[i - 96];
    else v = bhh[i - 192];
    sBias[i] = v;
  }
  if (tid < 128) sMu[tid] = slot_mu[tid];
  __syncthreads();

  float* ks_ = sWave[wid];
  float* vs_ = ks_ + 512;
  float* sl_ = vs_ + 512;
  float* scr = sl_ + 128;   // x / q(0..127) attn(128..191) upd(192..319)
  float* st_ = scr + 512;   // mean[16], rstd[16]

  long b = (long)blockIdx.x * WAVES_PER_WG + wid;
  if (b >= Bn) b = Bn - 1;  // duplicate-work clamp keeps barriers uniform

  const int n = lane & 15;
  const int rowb = (lane >> 4) * 8;
  const v8f zero = {0.f, 0.f, 0.f, 0.f, 0.f, 0.f, 0.f, 0.f};

  // ---- load latent[b] (8 channels x 16 px) coalesced into scr ----
  ((float4*)scr)[lane] = ((const float4*)(latent + b * 128))[lane];
  __syncthreads();

  // ---- x = latent^T @ w_in^T + b_in  (K padded 8 -> 32 with zeros) ----
  v16bf ax = packA(lane, [&](int mm, int k) { return (k < 8) ? scr[k * 16 + mm] : 0.f; });
  v8f x0 = wmma_bf16(ax, loadB(sWB + 0 * 256, lane), zero);
  v8f x1 = wmma_bf16(ax, loadB(sWB + 1 * 256, lane), zero);
#pragma unroll
  for (int r = 0; r < 8; ++r) {
    scr[(rowb + r) * 32 + n]      = x0[r] + sBias[n];
    scr[(rowb + r) * 32 + 16 + n] = x1[r] + sBias[16 + n];
  }
  __syncthreads();

  // ---- LayerNorm stats per token row ----
  if (lane < 16) {
    float mu = 0.f;
#pragma unroll
    for (int d = 0; d < 32; ++d) mu += scr[lane * 32 + d];
    mu *= (1.f / 32.f);
    float va = 0.f;
#pragma unroll
    for (int d = 0; d < 32; ++d) { float t = scr[lane * 32 + d] - mu; va += t * t; }
    st_[lane] = mu;
    st_[16 + lane] = rsqrtf(va * (1.f / 32.f) + LN_EPS);
  }
  __syncthreads();

  v16bf aln = packA(lane, [&](int mm, int k) {
    return (scr[mm * 32 + k] - st_[mm]) * st_[16 + mm] * sBias[32 + k] + sBias[64 + k];
  });

  // ---- k, v projections ----
  v8f k0 = wmma_bf16(aln, loadB(sWB + 2 * 256, lane), zero);
  v8f k1 = wmma_bf16(aln, loadB(sWB + 3 * 256, lane), zero);
  v8f v0 = wmma_bf16(aln, loadB(sWB + 4 * 256, lane), zero);
  v8f v1 = wmma_bf16(aln, loadB(sWB + 5 * 256, lane), zero);
#pragma unroll
  for (int r = 0; r < 8; ++r) {
    ks_[(rowb + r) * 32 + n] = k0[r];      ks_[(rowb + r) * 32 + 16 + n] = k1[r];
    vs_[(rowb + r) * 32 + n] = v0[r];      vs_[(rowb + r) * 32 + 16 + n] = v1[r];
  }
  __syncthreads();

  // ---- slots in WMMA C-layout registers (rows>=4 are dead padding) ----
  v8f sc0, sc1;
#pragma unroll
  for (int r = 0; r < 8; ++r) {
    int mm = rowb + r;
    sc0[r] = (mm < 4) ? sMu[mm * 32 + n] : 0.f;
    sc1[r] = (mm < 4) ? sMu[mm * 32 + 16 + n] : 0.f;
  }

  for (int it = 0; it < 3; ++it) {
    // slots C-layout -> LDS (valid rows only)
    if (rowb == 0) {
#pragma unroll
      for (int r = 0; r < 4; ++r) { sl_[r * 32 + n] = sc0[r]; sl_[r * 32 + 16 + n] = sc1[r]; }
    }
    __syncthreads();

    v16bf asl = packA(lane, [&](int mm, int k) { return (mm < 4) ? sl_[mm * 32 + k] : 0.f; });

    // q = slots @ wq^T
    v8f q0 = wmma_bf16(asl, loadB(sWB + 6 * 256, lane), zero);
    v8f q1 = wmma_bf16(asl, loadB(sWB + 7 * 256, lane), zero);
    if (rowb == 0) {
#pragma unroll
      for (int r = 0; r < 4; ++r) { scr[r * 32 + n] = q0[r]; scr[r * 32 + 16 + n] = q1[r]; }
    }
    __syncthreads();

    // attn logits (4 slots x 16 tokens), each lane does 2 dot-32s
    {
      int s0 = lane >> 4;
      float d0 = 0.f, d1 = 0.f;
#pragma unroll
      for (int d = 0; d < 32; ++d) {
        float kk = ks_[n * 32 + d];
        d0 += scr[s0 * 32 + d] * kk;
        d1 += scr[(s0 + 2) * 32 + d] * kk;
      }
      scr[128 + s0 * 16 + n] = d0 * ATT_SCALE;
      scr[128 + (s0 + 2) * 16 + n] = d1 * ATT_SCALE;
    }
    __syncthreads();

    // softmax over the 4 slots per token
    if (lane < 16) {
      float a0 = scr[128 + lane], a1 = scr[144 + lane], a2 = scr[160 + lane], a3 = scr[176 + lane];
      float mx = fmaxf(fmaxf(a0, a1), fmaxf(a2, a3));
      float e0 = __expf(a0 - mx), e1 = __expf(a1 - mx), e2 = __expf(a2 - mx), e3 = __expf(a3 - mx);
      float inv = 1.f / (e0 + e1 + e2 + e3);
      scr[128 + lane] = e0 * inv; scr[144 + lane] = e1 * inv;
      scr[160 + lane] = e2 * inv; scr[176 + lane] = e3 * inv;
    }
    __syncthreads();

    // updates[s][d] = sum_n attn[s][n] * v[n][d]   (lane = d)
#pragma unroll
    for (int s = 0; s < 4; ++s) {
      float acc = 0.f;
#pragma unroll
      for (int t = 0; t < 16; ++t) acc += scr[128 + s * 16 + t] * vs_[t * 32 + lane];
      scr[192 + s * 32 + lane] = acc;
    }
    __syncthreads();

    // GRU: gi = upd @ wih^T, gh = slots @ whh^T  (6 output tiles each, 96 cols)
    v16bf aup = packA(lane, [&](int mm, int k) { return (mm < 4) ? scr[192 + mm * 32 + k] : 0.f; });
    v8f gi0 = wmma_bf16(aup, loadB(sWB +  8 * 256, lane), zero);
    v8f gi1 = wmma_bf16(aup, loadB(sWB +  9 * 256, lane), zero);
    v8f gi2 = wmma_bf16(aup, loadB(sWB + 10 * 256, lane), zero);
    v8f gi3 = wmma_bf16(aup, loadB(sWB + 11 * 256, lane), zero);
    v8f gi4 = wmma_bf16(aup, loadB(sWB + 12 * 256, lane), zero);
    v8f gi5 = wmma_bf16(aup, loadB(sWB + 13 * 256, lane), zero);
    v8f gh0 = wmma_bf16(asl, loadB(sWB + 14 * 256, lane), zero);
    v8f gh1 = wmma_bf16(asl, loadB(sWB + 15 * 256, lane), zero);
    v8f gh2 = wmma_bf16(asl, loadB(sWB + 16 * 256, lane), zero);
    v8f gh3 = wmma_bf16(asl, loadB(sWB + 17 * 256, lane), zero);
    v8f gh4 = wmma_bf16(asl, loadB(sWB + 18 * 256, lane), zero);
    v8f gh5 = wmma_bf16(asl, loadB(sWB + 19 * 256, lane), zero);

    // elementwise GRU fully in registers: gi/gh tiles share (row,col)->lane mapping
#pragma unroll
    for (int r = 0; r < 8; ++r) {
      float rl = sigm((gi0[r] + sBias[96 + n])      + (gh0[r] + sBias[192 + n]));
      float rh = sigm((gi1[r] + sBias[112 + n])     + (gh1[r] + sBias[208 + n]));
      float zl = sigm((gi2[r] + sBias[128 + n])     + (gh2[r] + sBias[224 + n]));
      float zh = sigm((gi3[r] + sBias[144 + n])     + (gh3[r] + sBias[240 + n]));
      float nl = tanhf((gi4[r] + sBias[160 + n]) + rl * (gh4[r] + sBias[256 + n]));
      float nh = tanhf((gi5[r] + sBias[176 + n]) + rh * (gh5[r] + sBias[272 + n]));
      sc0[r] = (1.f - zl) * nl + zl * sc0[r];
      sc1[r] = (1.f - zh) * nh + zh * sc1[r];
    }
    __syncthreads();
  }

  // ---- store slots [4][32] ----
  if (rowb == 0) {
    float* ob = out + b * 128;
#pragma unroll
    for (int r = 0; r < 4; ++r) {
      ob[r * 32 + n] = sc0[r];
      ob[r * 32 + 16 + n] = sc1[r];
    }
  }
}

extern "C" void kernel_launch(void* const* d_in, const int* in_sizes, int n_in,
                              void* d_out, int out_size, void* d_ws, size_t ws_size,
                              hipStream_t stream) {
  const float* latent  = (const float*)d_in[0];
  const float* slot_mu = (const float*)d_in[1];
  const float* w_in    = (const float*)d_in[2];
  const float* b_in    = (const float*)d_in[3];
  const float* ln_g    = (const float*)d_in[4];
  const float* ln_b    = (const float*)d_in[5];
  const float* wq      = (const float*)d_in[6];
  const float* wk      = (const float*)d_in[7];
  const float* wv      = (const float*)d_in[8];
  const float* wih     = (const float*)d_in[9];
  const float* whh     = (const float*)d_in[10];
  const float* bih     = (const float*)d_in[11];
  const float* bhh     = (const float*)d_in[12];
  float* out = (float*)d_out;

  int B = in_sizes[0] / 128;  // latent is [B, 8, 4, 4]
  int nblk = (B + WAVES_PER_WG - 1) / WAVES_PER_WG;
  slot_attn_kernel<<<nblk, 128, 0, stream>>>(latent, slot_mu, w_in, b_in, ln_g, ln_b,
                                             wq, wk, wv, wih, whh, bih, bhh, out, B);
}
